// _DiffPoolDecoder_36670430773837
// MI455X (gfx1250) — matrix operand order repaired and checked by reading, not compiled, not run on hardware
//
#include <hip/hip_runtime.h>
#include <hip/hip_bf16.h>
#include <math.h>

// ---------------- problem constants ----------------
#define BG   100
#define NN   1000
#define DEG  16
#define EDG  (BG * NN * DEG)     // 1,600,000
#define KP1  800
#define KP2  640
#define KP3  512
#define HF   128                 // F_IN == H == 128
#define OUTC 10

typedef __attribute__((ext_vector_type(16))) _Float16 v16h;
typedef __attribute__((ext_vector_type(8)))  float    v8f;

// ---------------- utility kernels ----------------
__global__ void k_zero_f32(float* __restrict__ p, size_t n) {
    size_t i = (size_t)blockIdx.x * blockDim.x + threadIdx.x;
    size_t stride = (size_t)gridDim.x * blockDim.x;
    for (; i < n; i += stride) p[i] = 0.0f;
}

__global__ void k_cvt_f16(const float* __restrict__ s, _Float16* __restrict__ d, int n) {
    int i = blockIdx.x * blockDim.x + threadIdx.x;
    if (i < n) d[i] = (_Float16)s[i];
}

// ---------------- edge scatter: agg[dst] += h[src] (masked) ----------------
// one wave32 per edge, 4 floats per lane
__global__ __launch_bounds__(256) void k_scatter_add(
    const float* __restrict__ h, const int* __restrict__ src,
    const int* __restrict__ dst, const int* __restrict__ emask,
    float* __restrict__ agg, int E)
{
    int wave = (blockIdx.x * blockDim.x + threadIdx.x) >> 5;
    int lane = threadIdx.x & 31;
    if (wave >= E) return;
    if (emask && emask[wave] == 0) return;
    int s = src[wave], d = dst[wave];
    const float* hp = h + (size_t)s * HF;
    float* ap = agg + (size_t)d * HF;
#pragma unroll
    for (int j = 0; j < 4; ++j) {
        int f = lane + 32 * j;
        atomicAdd(ap + f, hp[f]);
    }
}

// ---------------- WMMA graph-conv GEMM ----------------
// out = relu(agg @ Wrel^T + x @ Wroot^T + bias), all [M,128], W row-major [out][in]
// A fragment (16-bit 16x32): lane<16 row=lane, K={k0..k0+15}, k0 = kt*32
//                            lane>=16 same row=lane-16, K shifted by +16
static __device__ inline v16h load_a_frag(const float* __restrict__ p) {
    v16h a;
#pragma unroll
    for (int j = 0; j < 8; ++j) a[j]     = (_Float16)p[j];
#pragma unroll
    for (int j = 0; j < 8; ++j) a[8 + j] = (_Float16)p[8 + j];
    return a;
}

__global__ __launch_bounds__(256) void k_conv_gemm(
    const float* __restrict__ A,            // agg  [M,128]
    const float* __restrict__ X,            // root [M,128]
    const _Float16* __restrict__ Wrel,      // [128,128] f16
    const _Float16* __restrict__ Wroot,     // [128,128] f16
    const float* __restrict__ bias,         // [128]
    float* __restrict__ out, int M)
{
    const int lane   = threadIdx.x & 31;
    const int wv     = threadIdx.x >> 5;        // 8 waves -> 8 N tiles
    const int rowB   = blockIdx.x * 16;
    const int laneHi = lane >> 4;               // 0/1
    const int mrow   = rowB + (lane & 15);
    const int ncol   = wv * 16 + (lane & 15);

    v8f acc = {};
#pragma unroll
    for (int kt = 0; kt < 4; ++kt) {
        const int k0 = kt * 32 + laneHi * 16;   // A chunk base (16 contiguous)
        const int kb = kt * 32 + laneHi * 16;   // B chunk base (16 contiguous)

        v16h a_agg = load_a_frag(A + (size_t)mrow * HF + k0);
        v16h a_x   = load_a_frag(X + (size_t)mrow * HF + k0);
        // B[k][n] = W[n][k] -> lane holds 16 consecutive halves of W row ncol
        v16h b_rel  = *(const v16h*)(Wrel  + (size_t)ncol * HF + kb);
        v16h b_root = *(const v16h*)(Wroot + (size_t)ncol * HF + kb);

        acc = __builtin_amdgcn_wmma_f32_16x16x32_f16(
            false, a_agg, false, b_rel, (short)0, acc, false, false);
        acc = __builtin_amdgcn_wmma_f32_16x16x32_f16(
            false, a_x, false, b_root, (short)0, acc, false, false);
    }

    const float bn = bias[ncol];
    const int rbase = rowB + laneHi * 8;        // C/D: vgpr r -> row r (+8 hi half)
#pragma unroll
    for (int r = 0; r < 8; ++r) {
        float v = acc[r] + bn;
        out[(size_t)(rbase + r) * HF + ncol] = v > 0.0f ? v : 0.0f;
    }
}

// ---------------- pooling score: tanh(h.w / ||w||), one wave per node ----------------
__global__ __launch_bounds__(256) void k_score(
    const float* __restrict__ h, const float* __restrict__ pw,
    float* __restrict__ score, int M)
{
    int node = (blockIdx.x * blockDim.x + threadIdx.x) >> 5;
    int lane = threadIdx.x & 31;
    if (node >= M) return;
    float dot = 0.0f, nrm = 0.0f;
#pragma unroll
    for (int j = 0; j < 4; ++j) {
        float w = pw[lane + 32 * j];
        float v = h[(size_t)node * HF + lane + 32 * j];
        dot += w * v;
        nrm += w * w;
    }
#pragma unroll
    for (int off = 16; off > 0; off >>= 1) {
        dot += __shfl_xor(dot, off, 32);
        nrm += __shfl_xor(nrm, off, 32);
    }
    if (lane == 0) score[node] = tanhf(dot * rsqrtf(nrm));
}

// ---------------- per-graph exact top-k via LDS bitonic sort ----------------
#define NPAD 1024
__global__ __launch_bounds__(512) void k_topk(
    const float* __restrict__ score, int n, int k,
    int* __restrict__ perm, float* __restrict__ pvals, int* __restrict__ mapping)
{
    __shared__ float sv[NPAD];
    __shared__ int   si[NPAD];
    const int b = blockIdx.x, tid = threadIdx.x;

    for (int i = tid; i < NPAD; i += 512) {
        if (i < n) {
            sv[i] = score[b * n + i];
            si[i] = i;
            mapping[b * n + i] = -1;
        } else {
            sv[i] = -INFINITY;
            si[i] = NPAD;
        }
    }
    // sort descending by value, ascending by index on ties (matches lax.top_k)
    for (int kk = 2; kk <= NPAD; kk <<= 1) {
        for (int j = kk >> 1; j > 0; j >>= 1) {
            __syncthreads();
            for (int i = tid; i < NPAD; i += 512) {
                int ixj = i ^ j;
                if (ixj > i) {
                    float va = sv[i], vb = sv[ixj];
                    int   ia = si[i], ib = si[ixj];
                    bool bef = (va > vb) || (va == vb && ia < ib);
                    bool up  = ((i & kk) == 0);
                    if (up ? !bef : bef) {
                        sv[i] = vb; sv[ixj] = va;
                        si[i] = ib; si[ixj] = ia;
                    }
                }
            }
        }
    }
    __syncthreads();
    for (int t = tid; t < k; t += 512) {
        int g  = b * n + si[t];
        int ng = b * k + t;
        perm[ng]  = g;
        pvals[ng] = sv[t];
        mapping[g] = ng;
    }
}

// ---------------- gather + score-scale: new_x[j] = h[perm[j]] * vals[j] ----------------
__global__ __launch_bounds__(256) void k_gather_scale(
    const float* __restrict__ h, const int* __restrict__ perm,
    const float* __restrict__ pvals, float* __restrict__ out, int Mk)
{
    int node = (blockIdx.x * blockDim.x + threadIdx.x) >> 5;
    int lane = threadIdx.x & 31;
    if (node >= Mk) return;
    int g = perm[node];
    float v = pvals[node];
    const float* hp = h + (size_t)g * HF;
    float* op = out + (size_t)node * HF;
#pragma unroll
    for (int j = 0; j < 4; ++j) {
        int f = lane + 32 * j;
        op[f] = hp[f] * v;
    }
}

// ---------------- edge remap through pooling mapping ----------------
__global__ void k_edge_remap(
    const int* __restrict__ s_in, const int* __restrict__ d_in_,
    const int* __restrict__ m_in, const int* __restrict__ mapping,
    int* __restrict__ s_out, int* __restrict__ d_out_, int* __restrict__ m_out, int E)
{
    int e = blockIdx.x * blockDim.x + threadIdx.x;
    if (e >= E) return;
    int mm = m_in ? m_in[e] : 1;
    int ns = mapping[s_in[e]];
    int nd = mapping[d_in_[e]];
    int valid = (mm != 0) && (ns >= 0) && (nd >= 0);
    s_out[e]  = valid ? ns : 0;
    d_out_[e] = valid ? nd : 0;
    m_out[e]  = valid;
}

// ---------------- global max+mean pool, accumulated across layers ----------------
__global__ __launch_bounds__(128) void k_global_pool_acc(
    const float* __restrict__ h, float* __restrict__ gacc, int k)
{
    int b = blockIdx.x, f = threadIdx.x;
    const float* p = h + (size_t)b * k * HF + f;
    float mx = -3.402823466e38f, sm = 0.0f;
    for (int i = 0; i < k; ++i) {
        float v = p[(size_t)i * HF];
        mx = fmaxf(mx, v);
        sm += v;
    }
    gacc[b * 256 + f]       += mx;
    gacc[b * 256 + 128 + f] += sm / (float)k;
}

// ---------------- tiny MLP head ----------------
__global__ void k_linear_relu(
    const float* __restrict__ in, const float* __restrict__ w,
    const float* __restrict__ bias, float* __restrict__ out, int fin)
{
    int row = blockIdx.x, o = threadIdx.x;
    const float* ip = in + (size_t)row * fin;
    const float* wp = w + (size_t)o * fin;
    float s = bias[o];
    for (int kk = 0; kk < fin; ++kk) s += ip[kk] * wp[kk];
    out[(size_t)row * blockDim.x + o] = fmaxf(s, 0.0f);
}

__global__ __launch_bounds__(32) void k_final_logsoftmax(
    const float* __restrict__ in, const float* __restrict__ w,
    const float* __restrict__ bias, float* __restrict__ out)
{
    __shared__ float z[OUTC];
    __shared__ float red[2];
    int row = blockIdx.x, t = threadIdx.x;
    if (t < OUTC) {
        const float* ip = in + (size_t)row * 64;
        const float* wp = w + (size_t)t * 64;
        float s = bias[t];
        for (int kk = 0; kk < 64; ++kk) s += ip[kk] * wp[kk];
        z[t] = s;
    }
    __syncthreads();
    if (t == 0) {
        float m = z[0];
        for (int i = 1; i < OUTC; ++i) m = fmaxf(m, z[i]);
        float se = 0.0f;
        for (int i = 0; i < OUTC; ++i) se += __expf(z[i] - m);
        red[0] = m;
        red[1] = logf(se);
    }
    __syncthreads();
    if (t < OUTC) out[row * OUTC + t] = z[t] - red[0] - red[1];
}

// ---------------- orchestration ----------------
extern "C" void kernel_launch(void* const* d_in, const int* in_sizes, int n_in,
                              void* d_out, int out_size, void* d_ws, size_t ws_size,
                              hipStream_t stream) {
    const float* x       = (const float*)d_in[0];
    const int*   ei      = (const int*)d_in[1];   // [2, E]
    const float* w_root1 = (const float*)d_in[2];
    const float* w_rel1  = (const float*)d_in[3];
    const float* b1      = (const float*)d_in[4];
    const float* pw1     = (const float*)d_in[5];
    const float* w_root2 = (const float*)d_in[6];
    const float* w_rel2  = (const float*)d_in[7];
    const float* b2      = (const float*)d_in[8];
    const float* pw2     = (const float*)d_in[9];
    const float* w_root3 = (const float*)d_in[10];
    const float* w_rel3  = (const float*)d_in[11];
    const float* b3      = (const float*)d_in[12];
    const float* pw3     = (const float*)d_in[13];
    const float* lin1_w  = (const float*)d_in[14];
    const float* lin1_b  = (const float*)d_in[15];
    const float* lin2_w  = (const float*)d_in[16];
    const float* lin2_b  = (const float*)d_in[17];
    const float* lin3_w  = (const float*)d_in[18];
    const float* lin3_b  = (const float*)d_in[19];
    float* out = (float*)d_out;

    const int* src0 = ei;
    const int* dst0 = ei + EDG;

    // ---- carve workspace ----
    char* ws = (char*)d_ws;
    size_t off = 0;
    auto carve = [&](size_t bytes) -> void* {
        void* p = ws + off;
        off = (off + bytes + 255) & ~(size_t)255;
        return p;
    };
    _Float16* wrel1h  = (_Float16*)carve(HF * HF * 2);
    _Float16* wroot1h = (_Float16*)carve(HF * HF * 2);
    _Float16* wrel2h  = (_Float16*)carve(HF * HF * 2);
    _Float16* wroot2h = (_Float16*)carve(HF * HF * 2);
    _Float16* wrel3h  = (_Float16*)carve(HF * HF * 2);
    _Float16* wroot3h = (_Float16*)carve(HF * HF * 2);
    float* agg     = (float*)carve((size_t)BG * NN * HF * 4);   // 51.2 MB
    float* conv    = (float*)carve((size_t)BG * NN * HF * 4);   // 51.2 MB
    float* pooled  = (float*)carve((size_t)BG * KP1 * HF * 4);  // 41.0 MB
    float* score   = (float*)carve((size_t)BG * NN * 4);
    int*   perm    = (int*)carve((size_t)BG * KP1 * 4);
    float* pvals   = (float*)carve((size_t)BG * KP1 * 4);
    int*   mapping = (int*)carve((size_t)BG * NN * 4);
    int*   srcW    = (int*)carve((size_t)EDG * 4);
    int*   dstW    = (int*)carve((size_t)EDG * 4);
    int*   emW     = (int*)carve((size_t)EDG * 4);
    float* gacc    = (float*)carve((size_t)BG * 256 * 4);
    float* m1      = (float*)carve((size_t)BG * 128 * 4);
    float* m2      = (float*)carve((size_t)BG * 64 * 4);
    (void)ws_size; (void)in_sizes; (void)n_in; (void)out_size;

    // ---- weight conversion to f16 (weights stay hot in L2) ----
    {
        int n = HF * HF, g = (n + 255) / 256;
        k_cvt_f16<<<g, 256, 0, stream>>>(w_rel1, wrel1h, n);
        k_cvt_f16<<<g, 256, 0, stream>>>(w_root1, wroot1h, n);
        k_cvt_f16<<<g, 256, 0, stream>>>(w_rel2, wrel2h, n);
        k_cvt_f16<<<g, 256, 0, stream>>>(w_root2, wroot2h, n);
        k_cvt_f16<<<g, 256, 0, stream>>>(w_rel3, wrel3h, n);
        k_cvt_f16<<<g, 256, 0, stream>>>(w_root3, wroot3h, n);
    }
    k_zero_f32<<<64, 256, 0, stream>>>(gacc, (size_t)BG * 256);

    const int M1 = BG * NN;   // 100000
    const int M2 = BG * KP1;  // 80000
    const int M3 = BG * KP2;  // 64000
    const int M4 = BG * KP3;  // 51200

    // ================= layer 1 =================
    k_zero_f32<<<2048, 256, 0, stream>>>(agg, (size_t)M1 * HF);
    k_scatter_add<<<EDG / 8, 256, 0, stream>>>(x, src0, dst0, nullptr, agg, EDG);
    k_conv_gemm<<<M1 / 16, 256, 0, stream>>>(agg, x, wrel1h, wroot1h, b1, conv, M1);
    k_score<<<M1 / 8, 256, 0, stream>>>(conv, pw1, score, M1);
    k_topk<<<BG, 512, 0, stream>>>(score, NN, KP1, perm, pvals, mapping);
    k_gather_scale<<<M2 / 8, 256, 0, stream>>>(conv, perm, pvals, pooled, M2);
    k_edge_remap<<<(EDG + 255) / 256, 256, 0, stream>>>(src0, dst0, nullptr, mapping,
                                                        srcW, dstW, emW, EDG);
    k_global_pool_acc<<<BG, 128, 0, stream>>>(pooled, gacc, KP1);

    // ================= layer 2 =================
    k_zero_f32<<<2048, 256, 0, stream>>>(agg, (size_t)M2 * HF);
    k_scatter_add<<<EDG / 8, 256, 0, stream>>>(pooled, srcW, dstW, emW, agg, EDG);
    k_conv_gemm<<<M2 / 16, 256, 0, stream>>>(agg, pooled, wrel2h, wroot2h, b2, conv, M2);
    k_score<<<M2 / 8, 256, 0, stream>>>(conv, pw2, score, M2);
    k_topk<<<BG, 512, 0, stream>>>(score, KP1, KP2, perm, pvals, mapping);
    k_gather_scale<<<M3 / 8, 256, 0, stream>>>(conv, perm, pvals, pooled, M3);
    k_edge_remap<<<(EDG + 255) / 256, 256, 0, stream>>>(srcW, dstW, emW, mapping,
                                                        srcW, dstW, emW, EDG);
    k_global_pool_acc<<<BG, 128, 0, stream>>>(pooled, gacc, KP2);

    // ================= layer 3 =================
    k_zero_f32<<<2048, 256, 0, stream>>>(agg, (size_t)M3 * HF);
    k_scatter_add<<<EDG / 8, 256, 0, stream>>>(pooled, srcW, dstW, emW, agg, EDG);
    k_conv_gemm<<<M3 / 16, 256, 0, stream>>>(agg, pooled, wrel3h, wroot3h, b3, conv, M3);
    k_score<<<M3 / 8, 256, 0, stream>>>(conv, pw3, score, M3);
    k_topk<<<BG, 512, 0, stream>>>(score, KP2, KP3, perm, pvals, mapping);
    k_gather_scale<<<M4 / 8, 256, 0, stream>>>(conv, perm, pvals, pooled, M4);
    k_global_pool_acc<<<BG, 128, 0, stream>>>(pooled, gacc, KP3);

    // ================= MLP head =================
    k_linear_relu<<<BG, 128, 0, stream>>>(gacc, lin1_w, lin1_b, m1, 256);
    k_linear_relu<<<BG, 64, 0, stream>>>(m1, lin2_w, lin2_b, m2, 128);
    k_final_logsoftmax<<<BG, 32, 0, stream>>>(m2, lin3_w, lin3_b, out);
}
